// DGLFRM_22325240004646
// MI455X (gfx1250) — matrix operand order repaired and checked by reading, not compiled, not run on hardware
//
#include <hip/hip_runtime.h>
#include <math.h>

typedef _Float16 half_t;
typedef __attribute__((ext_vector_type(16))) _Float16 v16h_t;
typedef __attribute__((ext_vector_type(8)))  float    v8f_t;

#define N_NODES 8192
#define D_FEAT  1024
#define H_DIM   256
#define K_DIM   64
#define HD_DIM  32
#define EPSF    1e-7f

// ---------------- WMMA helpers (wave32, 16x16x32 f16 -> f32) ----------------

__device__ __forceinline__ v8f_t wmma16(v16h_t a, v16h_t b, v8f_t c) {
  return __builtin_amdgcn_wmma_f32_16x16x32_f16(
      /*neg_a=*/false, a, /*neg_b=*/false, b,
      /*c_mod=*/(short)0, c, /*reuse_a=*/false, /*reuse_b=*/false);
}

// A fragment (16x32, M x K) from row-major f16 [*, ld]; base at (m0,k0).
__device__ __forceinline__ v16h_t afrag_h(const half_t* base, int ld, int lane) {
  int r = lane & 15;
  int s = (lane >> 4) << 3;
  const half_t* p = base + (size_t)r * ld + s;
  v16h_t a;
  ((uint4*)&a)[0] = *(const uint4*)p;
  ((uint4*)&a)[1] = *(const uint4*)(p + 16);
  return a;
}

// A fragment from row-major f32 source, converting to f16 in-flight (fallback path).
__device__ __forceinline__ v16h_t afrag_f32(const float* base, int ld, int lane) {
  int r = lane & 15;
  int s = (lane >> 4) << 3;
  const float* p = base + (size_t)r * ld + s;
  v16h_t a;
#pragma unroll
  for (int e = 0; e < 8; ++e) {
    a[e]     = (half_t)p[e];
    a[e + 8] = (half_t)p[e + 16];
  }
  return a;
}

// B fragment (32x16, K x N) from TRANSPOSED row-major Bt[N][K]; base at (n0,k0).
__device__ __forceinline__ v16h_t bfrag_h(const half_t* bt, int ld, int lane) {
  int n = lane & 15;
  int koff = (lane >> 4) << 3;
  const half_t* p = bt + (size_t)n * ld + koff;
  v16h_t b;
  ((uint4*)&b)[0] = *(const uint4*)p;
  ((uint4*)&b)[1] = *(const uint4*)(p + 16);
  return b;
}

// ---------------- Stage A ----------------
// Fast path: rowsum + exact f16 conversion of (adj + I) in one pass.

__global__ void k_deg_conv(const float* __restrict__ adj, float* __restrict__ dvec,
                           half_t* __restrict__ adjh) {
  __shared__ float red[256];
  int i = blockIdx.x;
  const float* row = adj + (size_t)i * N_NODES;
  half_t* orow = adjh + (size_t)i * N_NODES;
  float s = 0.f;
  for (int t = threadIdx.x; t < N_NODES; t += 256) {
    float v = row[t];
    s += v;
    orow[t] = (half_t)(v + (t == i ? 1.f : 0.f));   // {0,1,2}: exact in f16
  }
  red[threadIdx.x] = s;
  __syncthreads();
  for (int off = 128; off > 0; off >>= 1) {
    if (threadIdx.x < off) red[threadIdx.x] += red[threadIdx.x + off];
    __syncthreads();
  }
  if (threadIdx.x == 0) dvec[i] = rsqrtf(red[0] + 1.0f);
}

// Fallback: degrees only (adj kept fp32).
__global__ void k_degree(const float* __restrict__ adj, float* __restrict__ dvec) {
  __shared__ float red[256];
  int i = blockIdx.x;
  const float* row = adj + (size_t)i * N_NODES;
  float s = 0.f;
  for (int t = threadIdx.x; t < N_NODES; t += 256) s += row[t];
  red[threadIdx.x] = s;
  __syncthreads();
  for (int off = 128; off > 0; off >>= 1) {
    if (threadIdx.x < off) red[threadIdx.x] += red[threadIdx.x + off];
    __syncthreads();
  }
  if (threadIdx.x == 0) dvec[i] = rsqrtf(red[0] + 1.0f);
}

// ---------------- Tiny transpose/convert preps ----------------

__global__ void k_prep_w1t(const float* __restrict__ W1, half_t* __restrict__ W1T) {
  int idx = blockIdx.x * 256 + threadIdx.x;   // D*H = 262144
  int h = idx >> 10, d = idx & 1023;
  W1T[(size_t)h * D_FEAT + d] = (half_t)W1[(size_t)d * H_DIM + h];
}

__global__ void k_prep_wcat(const float* __restrict__ Wpi, const float* __restrict__ Wm,
                            const float* __restrict__ Wls, half_t* __restrict__ WcatT) {
  int idx = blockIdx.x * 256 + threadIdx.x;   // 192*256 = 49152
  int c = idx >> 8, k = idx & 255;
  const float* W = (c < 64) ? Wpi : (c < 128) ? Wm : Wls;
  WcatT[(size_t)c * H_DIM + k] = (half_t)W[(size_t)k * K_DIM + (c & 63)];
}

__global__ void k_prep_wxt(const float* __restrict__ Wx, half_t* __restrict__ WxT) {
  int idx = blockIdx.x * 256 + threadIdx.x;   // 1024*64 = 65536
  int n = idx >> 6, k = idx & 63;
  WxT[(size_t)n * K_DIM + k] = (half_t)Wx[(size_t)k * D_FEAT + n];
}

// ---------------- Stage B: XWsT[c][j] = f16(dvec[j]*(x@W1 + b1)[j][c]) ----------------

__global__ void k_gemm_xw(const float* __restrict__ x, const half_t* __restrict__ W1T,
                          const float* __restrict__ b1, const float* __restrict__ dvec,
                          half_t* __restrict__ XWsT) {
  int lane = threadIdx.x & 31, wid = threadIdx.x >> 5;
  int m0 = blockIdx.x * 64 + (wid & 3) * 16;
  int n0 = (wid >> 2) * 128;
  v8f_t acc[8];
#pragma unroll
  for (int f = 0; f < 8; ++f)
#pragma unroll
    for (int e = 0; e < 8; ++e) acc[f][e] = 0.f;

  for (int k0 = 0; k0 < D_FEAT; k0 += 32) {
    v16h_t bfr[8];
#pragma unroll
    for (int f = 0; f < 8; ++f)
      bfr[f] = bfrag_h(W1T + (size_t)(n0 + f * 16) * D_FEAT + k0, D_FEAT, lane);
    v16h_t a = afrag_f32(x + (size_t)m0 * D_FEAT + k0, D_FEAT, lane);
#pragma unroll
    for (int f = 0; f < 8; ++f) acc[f] = wmma16(a, bfr[f], acc[f]);
  }
  int nn = lane & 15, mbase = m0 + (lane >> 4) * 8;
#pragma unroll
  for (int f = 0; f < 8; ++f) {
    int c = n0 + f * 16 + nn;
    float bias = b1[c];
    union { half_t h[8]; uint4 u; } tmp;
#pragma unroll
    for (int vi = 0; vi < 8; ++vi)
      tmp.h[vi] = (half_t)(dvec[mbase + vi] * (acc[f][vi] + bias));
    *(uint4*)(XWsT + (size_t)c * N_NODES + mbase) = tmp.u;
  }
}

// ---------------- Stage C (fast): h = leaky(dvec[i] * (adjh @ XWs)) ----------------
// adjh already contains +I, so no self-term epilogue.

__global__ void k_gemm_adj1_h(const half_t* __restrict__ adjh, const half_t* __restrict__ XWsT,
                              const float* __restrict__ dvec, half_t* __restrict__ hbf) {
  int lane = threadIdx.x & 31, wid = threadIdx.x >> 5;
  int m0 = blockIdx.x * 64 + (wid & 3) * 16;
  int n0 = (wid >> 2) * 128;
  v8f_t acc[8];
#pragma unroll
  for (int f = 0; f < 8; ++f)
#pragma unroll
    for (int e = 0; e < 8; ++e) acc[f][e] = 0.f;

  const half_t* arow = adjh + (size_t)m0 * N_NODES;
  for (int k0 = 0; k0 < N_NODES; k0 += 32) {
    __builtin_prefetch(arow + (size_t)(lane & 15) * N_NODES + k0 + 1024, 0, 1);
    v16h_t bfr[8];
#pragma unroll
    for (int f = 0; f < 8; ++f)
      bfr[f] = bfrag_h(XWsT + (size_t)(n0 + f * 16) * N_NODES + k0, N_NODES, lane);
    v16h_t a = afrag_h(arow + k0, N_NODES, lane);
#pragma unroll
    for (int f = 0; f < 8; ++f) acc[f] = wmma16(a, bfr[f], acc[f]);
  }
  int nn = lane & 15, mh = (lane >> 4) * 8;
#pragma unroll
  for (int f = 0; f < 8; ++f) {
    int c = n0 + f * 16 + nn;
#pragma unroll
    for (int vi = 0; vi < 8; ++vi) {
      int m = m0 + mh + vi;
      float v = dvec[m] * acc[f][vi];
      hbf[(size_t)m * H_DIM + c] = (half_t)(v > 0.f ? v : 0.01f * v);
    }
  }
}

// Stage C (fallback): A from fp32 adj, identity via epilogue self-term.
__global__ void k_gemm_adj1(const float* __restrict__ adj, const half_t* __restrict__ XWsT,
                            const float* __restrict__ dvec, half_t* __restrict__ hbf) {
  int lane = threadIdx.x & 31, wid = threadIdx.x >> 5;
  int m0 = blockIdx.x * 64 + (wid & 3) * 16;
  int n0 = (wid >> 2) * 128;
  v8f_t acc[8];
#pragma unroll
  for (int f = 0; f < 8; ++f)
#pragma unroll
    for (int e = 0; e < 8; ++e) acc[f][e] = 0.f;

  const float* arow = adj + (size_t)m0 * N_NODES;
  for (int k0 = 0; k0 < N_NODES; k0 += 32) {
    __builtin_prefetch(arow + (size_t)(lane & 15) * N_NODES + k0 + 512, 0, 1);
    v16h_t bfr[8];
#pragma unroll
    for (int f = 0; f < 8; ++f)
      bfr[f] = bfrag_h(XWsT + (size_t)(n0 + f * 16) * N_NODES + k0, N_NODES, lane);
    v16h_t a = afrag_f32(arow + k0, N_NODES, lane);
#pragma unroll
    for (int f = 0; f < 8; ++f) acc[f] = wmma16(a, bfr[f], acc[f]);
  }
  int nn = lane & 15, mh = (lane >> 4) * 8;
#pragma unroll
  for (int f = 0; f < 8; ++f) {
    int c = n0 + f * 16 + nn;
#pragma unroll
    for (int vi = 0; vi < 8; ++vi) {
      int m = m0 + mh + vi;
      float selfv = (float)XWsT[(size_t)c * N_NODES + m];
      float v = dvec[m] * (acc[f][vi] + selfv);
      hbf[(size_t)m * H_DIM + c] = (half_t)(v > 0.f ? v : 0.01f * v);
    }
  }
}

// ---------------- Stage D: tsT[c][j] = f16(dvec[j]*(h@Wcat + bcat)[j][c]) ----------------

__global__ void k_gemm_t(const half_t* __restrict__ hbf, const half_t* __restrict__ WcatT,
                         const float* __restrict__ bpi, const float* __restrict__ bm,
                         const float* __restrict__ bls, const float* __restrict__ dvec,
                         half_t* __restrict__ tsT) {
  int lane = threadIdx.x & 31, wid = threadIdx.x >> 5;
  int m0 = blockIdx.x * 64 + (wid & 3) * 16;
  int n0 = (wid >> 2) * 96;
  v8f_t acc[6];
#pragma unroll
  for (int f = 0; f < 6; ++f)
#pragma unroll
    for (int e = 0; e < 8; ++e) acc[f][e] = 0.f;

  for (int k0 = 0; k0 < H_DIM; k0 += 32) {
    v16h_t bfr[6];
#pragma unroll
    for (int f = 0; f < 6; ++f)
      bfr[f] = bfrag_h(WcatT + (size_t)(n0 + f * 16) * H_DIM + k0, H_DIM, lane);
    v16h_t a = afrag_h(hbf + (size_t)m0 * H_DIM + k0, H_DIM, lane);
#pragma unroll
    for (int f = 0; f < 6; ++f) acc[f] = wmma16(a, bfr[f], acc[f]);
  }
  int nn = lane & 15, mbase = m0 + (lane >> 4) * 8;
#pragma unroll
  for (int f = 0; f < 6; ++f) {
    int c = n0 + f * 16 + nn;
    float bias = (c < 64) ? bpi[c] : (c < 128) ? bm[c - 64] : bls[c - 128];
    union { half_t h[8]; uint4 u; } tmp;
#pragma unroll
    for (int vi = 0; vi < 8; ++vi)
      tmp.h[vi] = (half_t)(dvec[mbase + vi] * (acc[f][vi] + bias));
    *(uint4*)(tsT + (size_t)c * N_NODES + mbase) = tmp.u;
  }
}

// ---------------- Stage E (fast): plr[i][c] = dvec[i] * (adjh @ ts)[i][c] ----------------

__global__ void k_gemm_adj2_h(const half_t* __restrict__ adjh, const half_t* __restrict__ tsT,
                              const float* __restrict__ dvec, float* __restrict__ plr) {
  int lane = threadIdx.x & 31, wid = threadIdx.x >> 5;
  int m0 = blockIdx.x * 64 + (wid & 3) * 16;
  int n0 = (wid >> 2) * 96;
  v8f_t acc[6];
#pragma unroll
  for (int f = 0; f < 6; ++f)
#pragma unroll
    for (int e = 0; e < 8; ++e) acc[f][e] = 0.f;

  const half_t* arow = adjh + (size_t)m0 * N_NODES;
  for (int k0 = 0; k0 < N_NODES; k0 += 32) {
    __builtin_prefetch(arow + (size_t)(lane & 15) * N_NODES + k0 + 1024, 0, 1);
    v16h_t bfr[6];
#pragma unroll
    for (int f = 0; f < 6; ++f)
      bfr[f] = bfrag_h(tsT + (size_t)(n0 + f * 16) * N_NODES + k0, N_NODES, lane);
    v16h_t a = afrag_h(arow + k0, N_NODES, lane);
#pragma unroll
    for (int f = 0; f < 6; ++f) acc[f] = wmma16(a, bfr[f], acc[f]);
  }
  int nn = lane & 15, mh = (lane >> 4) * 8;
#pragma unroll
  for (int f = 0; f < 6; ++f) {
    int c = n0 + f * 16 + nn;
#pragma unroll
    for (int vi = 0; vi < 8; ++vi) {
      int m = m0 + mh + vi;
      plr[(size_t)m * 192 + c] = dvec[m] * acc[f][vi];
    }
  }
}

// Stage E (fallback).
__global__ void k_gemm_adj2(const float* __restrict__ adj, const half_t* __restrict__ tsT,
                            const float* __restrict__ dvec, float* __restrict__ plr) {
  int lane = threadIdx.x & 31, wid = threadIdx.x >> 5;
  int m0 = blockIdx.x * 64 + (wid & 3) * 16;
  int n0 = (wid >> 2) * 96;
  v8f_t acc[6];
#pragma unroll
  for (int f = 0; f < 6; ++f)
#pragma unroll
    for (int e = 0; e < 8; ++e) acc[f][e] = 0.f;

  const float* arow = adj + (size_t)m0 * N_NODES;
  for (int k0 = 0; k0 < N_NODES; k0 += 32) {
    __builtin_prefetch(arow + (size_t)(lane & 15) * N_NODES + k0 + 512, 0, 1);
    v16h_t bfr[6];
#pragma unroll
    for (int f = 0; f < 6; ++f)
      bfr[f] = bfrag_h(tsT + (size_t)(n0 + f * 16) * N_NODES + k0, N_NODES, lane);
    v16h_t a = afrag_f32(arow + k0, N_NODES, lane);
#pragma unroll
    for (int f = 0; f < 6; ++f) acc[f] = wmma16(a, bfr[f], acc[f]);
  }
  int nn = lane & 15, mh = (lane >> 4) * 8;
#pragma unroll
  for (int f = 0; f < 6; ++f) {
    int c = n0 + f * 16 + nn;
#pragma unroll
    for (int vi = 0; vi < 8; ++vi) {
      int m = m0 + mh + vi;
      float selfv = (float)tsT[(size_t)c * N_NODES + m];
      plr[(size_t)m * 192 + c] = dvec[m] * (acc[f][vi] + selfv);
    }
  }
}

// ---------------- Stage F: stick-breaking + concrete + gaussian -> z (f16) ----------------

__global__ void k_z(const float* __restrict__ plr, const float* __restrict__ u_beta,
                    const float* __restrict__ u_bern, const float* __restrict__ eps_r,
                    const float* __restrict__ beta_a, const float* __restrict__ beta_b,
                    half_t* __restrict__ zb) {
  int i = blockIdx.x * 256 + threadIdx.x;
  if (i >= N_NODES) return;
  float cs = 0.f;
  for (int k = 0; k < K_DIM; ++k) {
    float a  = log1pf(expf(beta_a[k]));                 // softplus
    float bb = log1pf(expf(beta_b[k]));
    float u  = u_beta[(size_t)i * K_DIM + k];
    float v  = powf(1.f - powf(u, 1.f / bb), 1.f / a);
    cs += logf(v + EPSF);
    float p = expf(cs);
    float prior = logf(p + EPSF) - logf(1.f - p + EPSF);
    float logit = plr[(size_t)i * 192 + k] + prior;
    float ub = u_bern[(size_t)i * K_DIM + k];
    float y = logit + logf(ub + EPSF) - logf(1.f - ub + EPSF);   // TEMP_POST == 1
    float gate = 1.f / (1.f + expf(-y));
    float r = plr[(size_t)i * 192 + 64 + k] +
              expf(plr[(size_t)i * 192 + 128 + k]) * eps_r[(size_t)i * K_DIM + k];
    zb[(size_t)i * K_DIM + k] = (half_t)(gate * r);
  }
}

// ---------------- Stage G: x_hat = z @ Wx + bx (fp32 out) ----------------

__global__ void k_gemm_xhat(const half_t* __restrict__ zb, const half_t* __restrict__ WxT,
                            const float* __restrict__ bx, float* __restrict__ out) {
  int lane = threadIdx.x & 31, wid = threadIdx.x >> 5;
  int m0 = blockIdx.x * 64 + (wid & 3) * 16;
  int n0 = blockIdx.y * 256 + (wid >> 2) * 128;
  v8f_t acc[8];
#pragma unroll
  for (int f = 0; f < 8; ++f)
#pragma unroll
    for (int e = 0; e < 8; ++e) acc[f][e] = 0.f;

#pragma unroll
  for (int k0 = 0; k0 < K_DIM; k0 += 32) {
    v16h_t bfr[8];
#pragma unroll
    for (int f = 0; f < 8; ++f)
      bfr[f] = bfrag_h(WxT + (size_t)(n0 + f * 16) * K_DIM + k0, K_DIM, lane);
    v16h_t a = afrag_h(zb + (size_t)m0 * K_DIM + k0, K_DIM, lane);
#pragma unroll
    for (int f = 0; f < 8; ++f) acc[f] = wmma16(a, bfr[f], acc[f]);
  }
  int nn = lane & 15, mh = (lane >> 4) * 8;
#pragma unroll
  for (int f = 0; f < 8; ++f) {
    int c = n0 + f * 16 + nn;
    float bias = bx[c];
#pragma unroll
    for (int vi = 0; vi < 8; ++vi)
      out[(size_t)(m0 + mh + vi) * D_FEAT + c] = acc[f][vi] + bias;
  }
}

// ---------------- Stage H: h2 = z @ Wd + bd (tiny, plain VALU) ----------------

__global__ void k_h2(const half_t* __restrict__ zb, const float* __restrict__ Wd,
                     const float* __restrict__ bd, half_t* __restrict__ h2b) {
  int tid = blockIdx.x * 256 + threadIdx.x;   // N*HD = 262144
  int i = tid >> 5, j = tid & 31;
  float s = bd[j];
  for (int k = 0; k < K_DIM; ++k)
    s += (float)zb[(size_t)i * K_DIM + k] * Wd[(size_t)k * HD_DIM + j];
  h2b[(size_t)i * HD_DIM + j] = (half_t)s;
}

// ---------------- Stage I: edge = h2 @ h2^T (K=32, one WMMA per tile) ----------------

__global__ void k_gemm_edge(const half_t* __restrict__ h2b, float* __restrict__ out) {
  int lane = threadIdx.x & 31, wid = threadIdx.x >> 5;
  int m0 = blockIdx.x * 64 + wid * 16;
  int n0 = blockIdx.y * 64;
  v8f_t acc[4];
#pragma unroll
  for (int f = 0; f < 4; ++f)
#pragma unroll
    for (int e = 0; e < 8; ++e) acc[f][e] = 0.f;

  v16h_t bfr[4];
#pragma unroll
  for (int f = 0; f < 4; ++f)
    bfr[f] = bfrag_h(h2b + (size_t)(n0 + f * 16) * HD_DIM, HD_DIM, lane);
  v16h_t a = afrag_h(h2b + (size_t)m0 * HD_DIM, HD_DIM, lane);
#pragma unroll
  for (int f = 0; f < 4; ++f) acc[f] = wmma16(a, bfr[f], acc[f]);

  int nn = lane & 15, mh = (lane >> 4) * 8;
#pragma unroll
  for (int f = 0; f < 4; ++f) {
    int c = n0 + f * 16 + nn;
#pragma unroll
    for (int vi = 0; vi < 8; ++vi)
      out[(size_t)(m0 + mh + vi) * N_NODES + c] = acc[f][vi];
  }
}

// ---------------- Launch ----------------

extern "C" void kernel_launch(void* const* d_in, const int* in_sizes, int n_in,
                              void* d_out, int out_size, void* d_ws, size_t ws_size,
                              hipStream_t stream) {
  const float* x      = (const float*)d_in[0];
  const float* adj    = (const float*)d_in[1];
  const float* W1     = (const float*)d_in[2];
  const float* b1     = (const float*)d_in[3];
  const float* Wpi    = (const float*)d_in[4];
  const float* bpi    = (const float*)d_in[5];
  const float* Wm     = (const float*)d_in[6];
  const float* bm     = (const float*)d_in[7];
  const float* Wls    = (const float*)d_in[8];
  const float* bls    = (const float*)d_in[9];
  const float* beta_a = (const float*)d_in[10];
  const float* beta_b = (const float*)d_in[11];
  const float* Wd     = (const float*)d_in[12];
  const float* bd     = (const float*)d_in[13];
  const float* Wx     = (const float*)d_in[14];
  const float* bx     = (const float*)d_in[15];
  const float* u_beta = (const float*)d_in[16];
  const float* u_bern = (const float*)d_in[17];
  const float* eps_r  = (const float*)d_in[18];

  char* ws = (char*)d_ws;
  size_t off = 0;
  auto carve = [&](size_t bytes) -> void* {
    void* p = ws + off;
    off = (off + bytes + 255) & ~(size_t)255;
    return p;
  };
  float*  dvec  = (float*) carve((size_t)N_NODES * 4);
  half_t* W1T   = (half_t*)carve((size_t)H_DIM * D_FEAT * 2);
  half_t* XWsT  = (half_t*)carve((size_t)H_DIM * N_NODES * 2);
  half_t* hbf   = (half_t*)carve((size_t)N_NODES * H_DIM * 2);
  half_t* WcatT = (half_t*)carve((size_t)192 * H_DIM * 2);
  half_t* tsT   = (half_t*)carve((size_t)192 * N_NODES * 2);
  float*  plr   = (float*) carve((size_t)N_NODES * 192 * 4);
  half_t* zb    = (half_t*)carve((size_t)N_NODES * K_DIM * 2);
  half_t* WxT   = (half_t*)carve((size_t)D_FEAT * K_DIM * 2);
  half_t* h2b   = (half_t*)carve((size_t)N_NODES * HD_DIM * 2);
  half_t* adjh  = (half_t*)carve((size_t)N_NODES * N_NODES * 2);  // 128 MB, optional
  const bool fast = (off <= ws_size);   // deterministic: depends only on ws_size

  float* xhat_out = (float*)d_out;
  float* edge_out = xhat_out + (size_t)N_NODES * D_FEAT;

  if (fast) k_deg_conv<<<N_NODES, 256, 0, stream>>>(adj, dvec, adjh);
  else      k_degree  <<<N_NODES, 256, 0, stream>>>(adj, dvec);

  k_prep_w1t <<<1024, 256, 0, stream>>>(W1, W1T);
  k_prep_wcat<<<192,  256, 0, stream>>>(Wpi, Wm, Wls, WcatT);
  k_prep_wxt <<<256,  256, 0, stream>>>(Wx, WxT);

  k_gemm_xw<<<128, 256, 0, stream>>>(x, W1T, b1, dvec, XWsT);

  if (fast) k_gemm_adj1_h<<<128, 256, 0, stream>>>(adjh, XWsT, dvec, hbf);
  else      k_gemm_adj1  <<<128, 256, 0, stream>>>(adj,  XWsT, dvec, hbf);

  k_gemm_t<<<128, 256, 0, stream>>>(hbf, WcatT, bpi, bm, bls, dvec, tsT);

  if (fast) k_gemm_adj2_h<<<128, 256, 0, stream>>>(adjh, tsT, dvec, plr);
  else      k_gemm_adj2  <<<128, 256, 0, stream>>>(adj,  tsT, dvec, plr);

  k_z <<<N_NODES / 256, 256, 0, stream>>>(plr, u_beta, u_bern, eps_r, beta_a, beta_b, zb);
  k_h2<<<1024, 256, 0, stream>>>(zb, Wd, bd, h2b);

  k_gemm_xhat<<<dim3(128, 4),   256, 0, stream>>>(zb, WxT, bx, xhat_out);
  k_gemm_edge<<<dim3(128, 128), 128, 0, stream>>>(h2b, edge_out);
}
